// Renet_81544249082103
// MI455X (gfx1250) — matrix operand order repaired and checked by reading, not compiled, not run on hardware
//
#include <hip/hip_runtime.h>
#include <stdint.h>
#include <stddef.h>

// ---------------- types ----------------
typedef __attribute__((ext_vector_type(16))) __bf16 v16bf;
typedef __attribute__((ext_vector_type(8)))  float  v8f;
typedef __attribute__((ext_vector_type(4)))  unsigned int u32x4;
typedef __attribute__((ext_vector_type(4)))  float  f32x4;

union FragBF { v16bf v; u32x4 q[2]; };
union AccSt  { v8f v; f32x4 f4[2]; };

// ---------------- problem dims ----------------
#define B_    32
#define C_    128
#define T_    64            // H == W == 64, sequence length
#define NROW  2048          // B_*64 sequences per LSTM
#define HSZ   16777216      // elements per h / xv / xh buffer (2048*64*128)

// ---------------- LSTM kernel LDS layout (bytes) ----------------
#define WROW_STRIDE 528                   // 256 bf16 + 8 pad  -> 4-bank lane stride
#define WCAT_LDS    (512*WROW_STRIDE)     // 270336
#define XROW_STRIDE 272                   // 128 bf16 + 8 pad
#define XS_BYTES    (64*XROW_STRIDE)      // 17408
#define XS0_OFF     WCAT_LDS
#define XS1_OFF     (XS0_OFF + XS_BYTES)
#define HS_OFF      (XS1_OFF + XS_BYTES)
#define LSTM_LDS    (HS_OFF + XS_BYTES)   // 322560 <= 320KB WGP LDS
#define CONV_LDS    (128*XROW_STRIDE)     // 34816

// ---------------- fast math ----------------
__device__ __forceinline__ float rcp_fast(float x)     { return __builtin_amdgcn_rcpf(x); }
__device__ __forceinline__ float sigmoid_fast(float x) { return rcp_fast(1.0f + __expf(-x)); }
__device__ __forceinline__ float tanh_fast(float x)    { float e = __expf(2.0f*x); return 1.0f - 2.0f*rcp_fast(e + 1.0f); }

// ---------------- CDNA5 async global->LDS (ASYNCcnt) ----------------
__device__ __forceinline__ void async_g2l_b128(unsigned lds_byte_addr, const char* gaddr) {
    asm volatile("global_load_async_to_lds_b128 %0, %1, off"
                 :: "v"(lds_byte_addr), "v"(gaddr) : "memory");
}
__device__ __forceinline__ void wait_asynccnt0() {
    asm volatile("s_wait_asynccnt 0" ::: "memory");
}

// =====================================================================
// Pass 1a: x [B,C,H,W] f32 -> bf16 in both scan layouts
//   xv[((b*64+h)*64+w)*128+c]   (vertical scan: seq over w)
//   xh[((b*64+w)*64+h)*128+c]   (horizontal scan: seq over h)
// =====================================================================
__global__ void pack_x_kernel(const float* __restrict__ x,
                              __bf16* __restrict__ xv, __bf16* __restrict__ xh) {
    int idx = blockIdx.x * 256 + threadIdx.x;          // 16,777,216 total
    float f = x[idx];
    int w = idx & 63, h = (idx >> 6) & 63, c = (idx >> 12) & 127, b = idx >> 19;
    __bf16 v = (__bf16)f;
    xv[(((b << 6) + h) * 64 + w) * 128 + c] = v;
    xh[(((b << 6) + w) * 64 + h) * 128 + c] = v;
}

// =====================================================================
// Pass 1b: weights -> bf16.  Wcat[l][n][k] = (k<128 ? Wih[n][k] : Whh[n][k-128])
// Row-major [512][256] is exactly the WMMA-B-friendly layout (B(K=k,N=n)=W[n][k]).
// convW[cid][o][c] = w_cid[o][c].
// =====================================================================
struct WPtrs { const float* wih[4]; const float* whh[4]; const float* cw[4]; };

__global__ void pack_w_kernel(WPtrs wp, __bf16* __restrict__ wcat, __bf16* __restrict__ convw) {
    int idx = blockIdx.x * 256 + threadIdx.x;          // 589,824 total
    if (idx < 4 * 512 * 256) {
        int l = idx >> 17;
        int r = idx & 131071;
        int n = r >> 8, k = r & 255;
        float v = (k < 128) ? wp.wih[l][n * 128 + k] : wp.whh[l][n * 128 + (k - 128)];
        wcat[idx] = (__bf16)v;
    } else {
        int j = idx - 4 * 512 * 256;
        if (j < 4 * 128 * 128) {
            int cid = j >> 14;
            convw[j] = (__bf16)wp.cw[cid][j & 16383];
        }
    }
}

// =====================================================================
// Pass 2: fused bidirectional LSTM scan.
// grid = (32 row-blocks, 4 lstms), block = 256 (8 waves).
// Per step: z[64x512] = [x_t | h_{t-1}](64x256) x Wcat^T(256x512) + b  via
// 1024 v_wmma_f32_16x16x32_bf16; wave w owns the i/f/g/o tiles of hidden
// units [16w,16w+16) so gate math stays in accumulator registers.
// =====================================================================
struct BiasPtrs { const float* b[4]; };

__device__ __forceinline__ void prefetch_x_tile(char* smem, int xsoff, const char* xbytes,
                                                int n0, int p, int wv, int ln) {
    const int lhalf = ln >> 4, lmod = ln & 15;
    #pragma unroll
    for (int j = 0; j < 4; ++j) {
        int chunk = (wv << 2) + j;                 // 0..31, each = 512B = 2 rows
        int row = (chunk << 1) + lhalf;
        int co  = lmod << 4;                       // 16B per lane
        unsigned lds_a = (unsigned)(size_t)(smem + xsoff + row * XROW_STRIDE + co);
        const char* g = xbytes + (size_t)(n0 + row) * 16384 + (size_t)p * 256 + co;
        async_g2l_b128(lds_a, g);
    }
}

__device__ __forceinline__ void copy_h_out(const char* smem, __bf16* hout,
                                           int n0, int p, int tid) {
    int row = tid >> 2, part = tid & 3;            // 4 threads per 256B row
    const u32x4* s = (const u32x4*)(smem + HS_OFF + row * XROW_STRIDE + part * 64);
    u32x4* d = (u32x4*)((char*)hout + (size_t)(n0 + row) * 16384 + (size_t)p * 256 + part * 64);
    #pragma unroll
    for (int q = 0; q < 4; ++q) d[q] = s[q];
}

__global__ __launch_bounds__(256, 1)
void lstm_scan_kernel(const __bf16* __restrict__ xv, const __bf16* __restrict__ xh,
                      __bf16* __restrict__ hbase, const __bf16* __restrict__ wcat_g,
                      BiasPtrs bp) {
    extern __shared__ u32x4 smem_v[];
    char* smem = (char*)smem_v;

    const int l     = blockIdx.y;                  // 0=vf 1=vb 2=hf 3=hb
    const int n0    = blockIdx.x * 64;
    const int tid   = threadIdx.x;
    const int wv    = tid >> 5;
    const int ln    = tid & 31;
    const int lhalf = ln >> 4;
    const int lmod  = ln & 15;
    const bool rev  = (l & 1);
    const char* xbytes = (const char*)((l < 2) ? xv : xh);
    __bf16* hout = hbase + (size_t)l * HSZ;

    // stage Wcat[l] into LDS (rows padded 512B -> 528B against bank conflicts)
    {
        const u32x4* src = (const u32x4*)((const char*)wcat_g + (size_t)l * 262144);
        #pragma unroll
        for (int rr = 0; rr < 2; ++rr) {
            int r = tid * 2 + rr;
            const u32x4* s = src + r * 32;
            u32x4* d = (u32x4*)(smem + r * WROW_STRIDE);
            #pragma unroll
            for (int q = 0; q < 32; ++q) d[q] = s[q];
        }
    }
    // zero h stage (h_{-1} = 0)
    {
        u32x4 z = {0u, 0u, 0u, 0u};
        u32x4* hs = (u32x4*)(smem + HS_OFF);
        for (int i = tid; i < XS_BYTES / 16; i += 256) hs[i] = z;
    }
    // per-lane biases for hidden unit u = 16*wave + lane%16
    const float* bptr = bp.b[l];
    const int u = (wv << 4) + lmod;
    float biasv[4];
    #pragma unroll
    for (int g = 0; g < 4; ++g) biasv[g] = bptr[g * C_ + u];

    // prefetch x @ p(0)
    prefetch_x_tile(smem, XS0_OFF, xbytes, n0, rev ? (T_ - 1) : 0, wv, ln);
    wait_asynccnt0();
    __syncthreads();

    v8f cst[4];
    #pragma unroll
    for (int m = 0; m < 4; ++m)
        #pragma unroll
        for (int e = 0; e < 8; ++e) cst[m][e] = 0.0f;

    for (int s = 0; s < T_; ++s) {
        const int p = rev ? (T_ - 1 - s) : s;
        const int buf = s & 1;
        if (s + 1 < T_)
            prefetch_x_tile(smem, buf ? XS0_OFF : XS1_OFF, xbytes, n0, rev ? p - 1 : p + 1, wv, ln);
        if (s > 0)
            copy_h_out(smem, hout, n0, rev ? p + 1 : p - 1, tid);

        // accumulators: [M-tile 0..3][gate i,f,g,o], init = bias
        v8f acc[4][4];
        #pragma unroll
        for (int m = 0; m < 4; ++m)
            #pragma unroll
            for (int g = 0; g < 4; ++g)
                #pragma unroll
                for (int e = 0; e < 8; ++e) acc[m][g][e] = biasv[g];

        // K loop: kc 0..3 from x stage (Wih cols), kc 4..7 from h stage (Whh cols)
        #pragma unroll
        for (int kc = 0; kc < 8; ++kc) {
            const int srcoff = (kc < 4) ? (buf ? XS1_OFF : XS0_OFF) : HS_OFF;
            const int kcol = (kc & 3) * 64;
            FragBF A[4];
            #pragma unroll
            for (int m = 0; m < 4; ++m) {
                const char* ab = smem + srcoff + ((m << 4) + lmod) * XROW_STRIDE + kcol + lhalf * 16;
                A[m].q[0] = *(const u32x4*)(ab);
                A[m].q[1] = *(const u32x4*)(ab + 32);
            }
            #pragma unroll
            for (int g = 0; g < 4; ++g) {
                const char* bb = smem + (size_t)((g << 7) + (wv << 4) + lmod) * WROW_STRIDE
                               + (kc << 6) + lhalf * 16;
                FragBF Bf;
                Bf.q[0] = *(const u32x4*)(bb);
                Bf.q[1] = *(const u32x4*)(bb + 32);
                #pragma unroll
                for (int m = 0; m < 4; ++m)
                    acc[m][g] = __builtin_amdgcn_wmma_f32_16x16x32_bf16(
                        false, A[m].v, false, Bf.v, (short)0, acc[m][g], false, false);
            }
        }

        // gate math fully in accumulator layout (f32)
        float hres[4][8];
        #pragma unroll
        for (int m = 0; m < 4; ++m)
            #pragma unroll
            for (int e = 0; e < 8; ++e) {
                float zi = acc[m][0][e], zf = acc[m][1][e], zg = acc[m][2][e], zo = acc[m][3][e];
                float c = sigmoid_fast(zf) * cst[m][e] + sigmoid_fast(zi) * tanh_fast(zg);
                cst[m][e] = c;
                hres[m][e] = sigmoid_fast(zo) * tanh_fast(c);
            }

        __syncthreads();   // all reads of h_{s-1} (WMMA + copy-out) are done
        #pragma unroll
        for (int m = 0; m < 4; ++m)
            #pragma unroll
            for (int e = 0; e < 8; ++e) {
                int row = (m << 4) + e + (lhalf << 3);
                *(__bf16*)(smem + HS_OFF + row * XROW_STRIDE + (u << 1)) = (__bf16)hres[m][e];
            }
        wait_asynccnt0();  // own async x(t+1) complete
        __syncthreads();   // publish h_s + all async data before next step
    }
    copy_h_out(smem, hout, n0, rev ? 0 : T_ - 1, tid);
}

// =====================================================================
// Pass 3: 1x1 conv = GEMM [131072 x 128] x [128 x 128] + bias -> f32 out.
// grid = (1024 M-blocks, 4 convs), block = 256; wave w owns M-tile w x all
// 8 N-tiles. Accumulators store as two contiguous float4 per lane.
// =====================================================================
struct ConvArgs { const float* bias[4]; };

__global__ __launch_bounds__(256, 1)
void conv1x1_kernel(const __bf16* __restrict__ hbase, const __bf16* __restrict__ convW,
                    ConvArgs ca, float* __restrict__ out) {
    extern __shared__ u32x4 smem_v[];
    char* smem = (char*)smem_v;

    const int cid = blockIdx.y;                    // 0=down 1=up 2=right 3=left
    const int lsel = (cid == 0) ? 1 : (cid == 1) ? 0 : cid;   // h_vb,h_vf,h_hf,h_hb
    const __bf16* hsrc = hbase + (size_t)lsel * HSZ;
    const char* hbytes = (const char*)hsrc;
    const int M0 = blockIdx.x * 128;
    const int tid = threadIdx.x, wv = tid >> 5, ln = tid & 31;
    const int lhalf = ln >> 4, lmod = ln & 15;

    // stage conv weight (128x128 bf16, padded rows)
    {
        int row = tid >> 1, half = tid & 1;
        const u32x4* s = (const u32x4*)((const char*)convW + (size_t)cid * 32768 + row * 256 + half * 128);
        u32x4* d = (u32x4*)(smem + row * XROW_STRIDE + half * 128);
        #pragma unroll
        for (int q = 0; q < 8; ++q) d[q] = s[q];
    }
    __syncthreads();

    const float* bptr = ca.bias[cid];
    v8f acc[8];
    #pragma unroll
    for (int nt = 0; nt < 8; ++nt) {
        float bv = bptr[(nt << 4) + lmod];
        #pragma unroll
        for (int e = 0; e < 8; ++e) acc[nt][e] = bv;
    }

    #pragma unroll
    for (int kc = 0; kc < 4; ++kc) {
        FragBF A;
        const char* ab = hbytes + (size_t)(M0 + (wv << 4) + lmod) * 256 + (kc << 6) + lhalf * 16;
        A.q[0] = *(const u32x4*)(ab);
        A.q[1] = *(const u32x4*)(ab + 32);
        #pragma unroll
        for (int nt = 0; nt < 8; ++nt) {
            const char* bb = smem + ((nt << 4) + lmod) * XROW_STRIDE + (kc << 6) + lhalf * 16;
            FragBF Bf;
            Bf.q[0] = *(const u32x4*)(bb);
            Bf.q[1] = *(const u32x4*)(bb + 32);
            acc[nt] = __builtin_amdgcn_wmma_f32_16x16x32_bf16(
                false, A.v, false, Bf.v, (short)0, acc[nt], false, false);
        }
    }

    // store: m = M0 + 16*wv + e + 8*lhalf -> out[sec][b][o][m%4096], contiguous in e
    const int m_base = M0 + (wv << 4) + (lhalf << 3);
    const int brow = m_base >> 12;
    const int rem = m_base & 4095;
    float* osec = out + (size_t)cid * 16777216;
    #pragma unroll
    for (int nt = 0; nt < 8; ++nt) {
        AccSt U; U.v = acc[nt];
        const int o = (nt << 4) + lmod;
        float* dst = osec + ((size_t)(brow * C_ + o) << 12) + rem;
        *(f32x4*)(dst) = U.f4[0];
        *(f32x4*)(dst + 4) = U.f4[1];
    }
}

// =====================================================================
// host launcher
// =====================================================================
extern "C" void kernel_launch(void* const* d_in, const int* in_sizes, int n_in,
                              void* d_out, int out_size, void* d_ws, size_t ws_size,
                              hipStream_t stream) {
    (void)in_sizes; (void)n_in; (void)out_size; (void)ws_size;
    const float* x = (const float*)d_in[0];

    // workspace layout (bytes): ~193 MB total
    char* ws = (char*)d_ws;
    __bf16* xv    = (__bf16*)(ws);                        //  33,554,432 B
    __bf16* xh    = (__bf16*)(ws + 33554432);             //  33,554,432 B
    __bf16* hbase = (__bf16*)(ws + 67108864);             // 4 x 33,554,432 B
    __bf16* wcat  = (__bf16*)(ws + 201326592);            //   1,048,576 B
    __bf16* convw = (__bf16*)(ws + 202375168);            //     131,072 B

    pack_x_kernel<<<65536, 256, 0, stream>>>(x, xv, xh);

    WPtrs wp;
    wp.wih[0] = (const float*)d_in[1];  wp.whh[0] = (const float*)d_in[2];
    wp.wih[1] = (const float*)d_in[4];  wp.whh[1] = (const float*)d_in[5];
    wp.wih[2] = (const float*)d_in[7];  wp.whh[2] = (const float*)d_in[8];
    wp.wih[3] = (const float*)d_in[10]; wp.whh[3] = (const float*)d_in[11];
    wp.cw[0] = (const float*)d_in[13];  // w_down
    wp.cw[1] = (const float*)d_in[15];  // w_up
    wp.cw[2] = (const float*)d_in[17];  // w_right
    wp.cw[3] = (const float*)d_in[19];  // w_left
    pack_w_kernel<<<2304, 256, 0, stream>>>(wp, wcat, convw);

    BiasPtrs bp;
    bp.b[0] = (const float*)d_in[3];    // vb_f
    bp.b[1] = (const float*)d_in[6];    // vb_b
    bp.b[2] = (const float*)d_in[9];    // hb_f
    bp.b[3] = (const float*)d_in[12];   // hb_b
    dim3 lg(32, 4);
    lstm_scan_kernel<<<lg, 256, LSTM_LDS, stream>>>(xv, xh, hbase, wcat, bp);

    ConvArgs ca;
    ca.bias[0] = (const float*)d_in[14];  // b_down
    ca.bias[1] = (const float*)d_in[16];  // b_up
    ca.bias[2] = (const float*)d_in[18];  // b_right
    ca.bias[3] = (const float*)d_in[20];  // b_left
    dim3 cg(1024, 4);
    conv1x1_kernel<<<cg, 256, CONV_LDS, stream>>>(hbase, convw, ca, (float*)d_out);
}
